// RelativeMultiheadAttention_31903017074724
// MI455X (gfx1250) — matrix operand order repaired and checked by reading, not compiled, not run on hardware
//
#include <hip/hip_runtime.h>

typedef __bf16 bf16_t;
typedef __bf16  v16bf __attribute__((ext_vector_type(16)));
typedef float   v8f   __attribute__((ext_vector_type(8)));

#define Bsz 4
#define Ssz 1024
#define Esz 512
#define Hsz 8
#define Dk  64
#define BH  (Bsz*Hsz)
#define NR  257      // 2*C+1 relative buckets
#define RP  288      // padded bucket stride (9 chunks of 32)

__device__ __forceinline__ bf16_t f2bf(float f) { return (bf16_t)f; }

__device__ __forceinline__ v8f wmma_bf16(v16bf a, v16bf b, v8f c) {
    return __builtin_amdgcn_wmma_f32_16x16x32_bf16(false, a, false, b, (short)0, c, false, false);
}

// =====================================================================
// K0: transpose a_v [257][64] f32 -> avt bf16 [64][RP] (zero padded)
// =====================================================================
__global__ __launch_bounds__(256) void k_avt(const float* __restrict__ a_v,
                                             bf16_t* __restrict__ avt) {
    int t = blockIdx.x * 256 + threadIdx.x;
    if (t >= Dk * RP) return;
    int v = t / RP, r = t % RP;
    float x = (r < NR) ? a_v[(size_t)r * Dk + v] : 0.0f;
    avt[t] = f2bf(x);
}

// =====================================================================
// K1: projection  out[bh] = outScale * (xs[b] @ W[h])   (S x 64, bf16)
// TRANSPOSE==false: out layout [BH][S][64]; true: out layout [BH][64][S]
// grid (BH, S/64), 256 thr = 8 waves; wave -> 16 rows x 32 cols
// Q is stored pre-scaled by 1/sqrt(dv)=0.125 (exact in bf16) so the
// attention kernels never multiply by the softmax scale.
// =====================================================================
template <bool TRANSPOSE>
__global__ __launch_bounds__(256) void k_proj(const float* __restrict__ xs,
                                              const float* __restrict__ w,
                                              bf16_t* __restrict__ out,
                                              float outScale) {
    int bh = blockIdx.x, b = bh / Hsz, h = bh % Hsz;
    int s0 = blockIdx.y * 64;
    int tid = threadIdx.x, lane = tid & 31, wv = tid >> 5;
    int lm = lane & 15, hf = lane >> 4;
    int rt = wv & 3, cp = wv >> 2;

    __shared__ bf16_t lx[64][72];   // xs tile  [row][k]
    __shared__ bf16_t lw[64][72];   // Wt tile  [n][k]

    v8f acc0 = 0.0f, acc1 = 0.0f;
    const float* xbase = xs + ((size_t)b * Ssz + s0) * Esz;
    const float* wbase = w + (size_t)h * Esz * Dk;

    for (int e0 = 0; e0 < Esz; e0 += 64) {
        // stage xs tile: vectorized float4 loads, 4 contiguous bf16 stores
        for (int t = tid; t < 64 * 16; t += 256) {
            int r = t >> 4, c4 = (t & 15) * 4;
            float4 x = *(const float4*)(xbase + (size_t)r * Esz + e0 + c4);
            lx[r][c4 + 0] = f2bf(x.x);
            lx[r][c4 + 1] = f2bf(x.y);
            lx[r][c4 + 2] = f2bf(x.z);
            lx[r][c4 + 3] = f2bf(x.w);
        }
        // stage W transposed: float4 loads along n, strided LDS writes
        for (int t = tid; t < 64 * 16; t += 256) {
            int kc = t >> 4, n4 = (t & 15) * 4;
            float4 x = *(const float4*)(wbase + (size_t)(e0 + kc) * Dk + n4);
            lw[n4 + 0][kc] = f2bf(x.x);
            lw[n4 + 1][kc] = f2bf(x.y);
            lw[n4 + 2][kc] = f2bf(x.z);
            lw[n4 + 3][kc] = f2bf(x.w);
        }
        __syncthreads();
        #pragma unroll
        for (int kk = 0; kk < 64; kk += 32) {
            int arow = rt * 16 + lm;
            int kb = hf * 8, kb2 = hf * 16;
            v16bf a;
            #pragma unroll
            for (int e = 0; e < 8; ++e) {
                a[e]     = lx[arow][kk + kb2 + e];
                a[8 + e] = lx[arow][kk + kb2 + 8 + e];
            }
            {
                int col = cp * 32 + lm;
                v16bf bb;
                #pragma unroll
                for (int e = 0; e < 16; ++e) bb[e] = lw[col][kk + kb2 + e];
                acc0 = wmma_bf16(a, bb, acc0);
            }
            {
                int col = cp * 32 + 16 + lm;
                v16bf bb;
                #pragma unroll
                for (int e = 0; e < 16; ++e) bb[e] = lw[col][kk + kb2 + e];
                acc1 = wmma_bf16(a, bb, acc1);
            }
        }
        __syncthreads();
    }
    // store D tiles (branch-free: layout is compile-time)
    #pragma unroll
    for (int t = 0; t < 2; ++t) {
        v8f acc = t ? acc1 : acc0;
        int col = cp * 32 + t * 16 + lm;
        #pragma unroll
        for (int r = 0; r < 8; ++r) {
            int row = s0 + rt * 16 + 8 * hf + r;
            bf16_t val = f2bf(acc[r] * outScale);
            if (TRANSPOSE) out[((size_t)bh * Dk + col) * Ssz + row] = val; // contiguous in r
            else           out[((size_t)bh * Ssz + row) * Dk + col] = val;
        }
    }
}

// =====================================================================
// K2: QA[bh][i][r] = sum_q Qscaled[i][q] * a_k[r][q]   (f32, RP stride)
// (Q already carries the 0.125 softmax scale.)
// a_k staged once into zero-padded LDS -> unconditional B fragments.
// grid (BH, S/64); wave -> 16 rows x 144 cols (9 tiles)
// =====================================================================
__global__ __launch_bounds__(256) void k_qa(const bf16_t* __restrict__ Qb,
                                            const float* __restrict__ a_k,
                                            float* __restrict__ QA) {
    int bh = blockIdx.x;
    int s0 = blockIdx.y * 64;
    int tid = threadIdx.x, lane = tid & 31, wv = tid >> 5;
    int lm = lane & 15, hf = lane >> 4;
    int rt = wv & 3, ch = wv >> 2;
    int kb = hf * 8, kb2 = hf * 16;

    __shared__ bf16_t lak[RP][Dk];   // 36 KB: a_k bf16, rows >= NR zeroed

    // stage a_k (row-uniform bounds check, float4 loads)
    for (int t = tid; t < RP * 16; t += 256) {
        int r = t >> 4, c4 = (t & 15) * 4;
        if (r < NR) {
            float4 x = *(const float4*)(a_k + (size_t)r * Dk + c4);
            lak[r][c4 + 0] = f2bf(x.x);
            lak[r][c4 + 1] = f2bf(x.y);
            lak[r][c4 + 2] = f2bf(x.z);
            lak[r][c4 + 3] = f2bf(x.w);
        } else {
            lak[r][c4 + 0] = f2bf(0.0f);
            lak[r][c4 + 1] = f2bf(0.0f);
            lak[r][c4 + 2] = f2bf(0.0f);
            lak[r][c4 + 3] = f2bf(0.0f);
        }
    }
    __syncthreads();

    v8f acc[9];
    #pragma unroll
    for (int t = 0; t < 9; ++t) acc[t] = 0.0f;

    int arow = s0 + rt * 16 + lm;
    const bf16_t* qrow = Qb + ((size_t)bh * Ssz + arow) * Dk;

    #pragma unroll
    for (int kk = 0; kk < 64; kk += 32) {
        v16bf a;
        #pragma unroll
        for (int e = 0; e < 8; ++e) {
            a[e]     = qrow[kk + kb2 + e];
            a[8 + e] = qrow[kk + kb2 + 8 + e];
        }
        #pragma unroll
        for (int t = 0; t < 9; ++t) {
            int r = (ch * 9 + t) * 16 + lm;
            v16bf bb;
            #pragma unroll
            for (int e = 0; e < 16; ++e) bb[e] = lak[r][kk + kb2 + e];
            acc[t] = wmma_bf16(a, bb, acc[t]);
        }
    }
    #pragma unroll
    for (int t = 0; t < 9; ++t) {
        int col = (ch * 9 + t) * 16 + lm;
        #pragma unroll
        for (int r = 0; r < 8; ++r) {
            int row = s0 + rt * 16 + 8 * hf + r;
            QA[((size_t)bh * Ssz + row) * RP + col] = acc[t][r];
        }
    }
}

// =====================================================================
// K3: fused attention per (bh, 32-row block)
// pass1: online row max/sum (per-wave j-stripes, merged via LDS)
// pass2: P (bf16->LDS), O += P*V via WMMA, bucket alphas, + wbuck@a_v
// =====================================================================
__global__ __launch_bounds__(256) void k_attn(const bf16_t* __restrict__ Qb,
                                              const bf16_t* __restrict__ Kb,
                                              const bf16_t* __restrict__ Vt,
                                              const float* __restrict__ QA,
                                              const bf16_t* __restrict__ avt,
                                              float* __restrict__ Zb) {
    int bh = blockIdx.x;
    int i0 = blockIdx.y * 32;
    int tid = threadIdx.x, lane = tid & 31, wv = tid >> 5;
    int lm = lane & 15, hf = lane >> 4;
    int rt = wv & 1;    // row tile (scores and O phases)
    int jt = wv >> 1;   // j subtile in scores phase / v tile in O phase
    int kb = hf * 8, kb2 = hf * 16;

    __shared__ float  wbuck[32][RP];        // 36 KB relative buckets
    __shared__ bf16_t Pl[32][64];           // 4 KB probability tile
    __shared__ float  mw[32][4], sw[32][4]; // per-wave stats
    __shared__ float  mrow[32], rsrow[32];

    for (int t = tid; t < 32 * RP; t += 256) (&wbuck[0][0])[t] = 0.0f;

    // hoist the (pre-scaled) Q A-fragments: constant across j
    int arow = i0 + rt * 16 + lm;
    const bf16_t* qrow = Qb + ((size_t)bh * Ssz + arow) * Dk;
    v16bf aq[2];
    #pragma unroll
    for (int kki = 0; kki < 2; ++kki) {
        int kk = kki * 32;
        #pragma unroll
        for (int e = 0; e < 8; ++e) {
            aq[kki][e]     = qrow[kk + kb2 + e];
            aq[kki][8 + e] = qrow[kk + kb2 + 8 + e];
        }
    }

    const bf16_t* kbase  = Kb + (size_t)bh * Ssz * Dk;
    const bf16_t* vtb    = Vt + (size_t)bh * Dk * Ssz;
    const float*  qabase = QA + (size_t)bh * Ssz * RP;

    float m_run[8], s_run[8];
    #pragma unroll
    for (int r = 0; r < 8; ++r) { m_run[r] = -3.0e38f; s_run[r] = 0.0f; }

    // ---------------- pass 1: row max & sum ----------------
    for (int j0 = 0; j0 < Ssz; j0 += 64) {
        int jb = j0 + jt * 16;
        int j = jb + lm;
        if (j0 + 64 < Ssz)
            __builtin_prefetch(kbase + (size_t)(j + 64) * Dk, 0, 1);
        v8f acc = 0.0f;
        #pragma unroll
        for (int kki = 0; kki < 2; ++kki) {
            const bf16_t* krow = kbase + (size_t)j * Dk + kki * 32;
            v16bf bb;
            #pragma unroll
            for (int e = 0; e < 16; ++e) bb[e] = krow[kb2 + e];
            acc = wmma_bf16(aq[kki], bb, acc);
        }
        float sc[8];
        #pragma unroll
        for (int r = 0; r < 8; ++r) {
            int i = i0 + rt * 16 + 8 * hf + r;
            int rel = j - i;
            rel = rel < -128 ? -128 : (rel > 128 ? 128 : rel);
            sc[r] = acc[r] + qabase[(size_t)i * RP + (rel + 128)];
        }
        #pragma unroll
        for (int r = 0; r < 8; ++r) {
            float tm = sc[r];
            #pragma unroll
            for (int off = 8; off >= 1; off >>= 1) tm = fmaxf(tm, __shfl_xor(tm, off, 32));
            float nm = fmaxf(m_run[r], tm);
            float ts = __expf(sc[r] - nm);
            #pragma unroll
            for (int off = 8; off >= 1; off >>= 1) ts += __shfl_xor(ts, off, 32);
            s_run[r] = s_run[r] * __expf(m_run[r] - nm) + ts;
            m_run[r] = nm;
        }
    }
    if (lm == 0) {
        #pragma unroll
        for (int r = 0; r < 8; ++r) {
            int row = rt * 16 + 8 * hf + r;
            mw[row][jt] = m_run[r]; sw[row][jt] = s_run[r];
        }
    }
    __syncthreads();
    if (tid < 32) {
        float m = mw[tid][0];
        #pragma unroll
        for (int t = 1; t < 4; ++t) m = fmaxf(m, mw[tid][t]);
        float s = 0.0f;
        #pragma unroll
        for (int t = 0; t < 4; ++t) s += sw[tid][t] * __expf(mw[tid][t] - m);
        mrow[tid] = m; rsrow[tid] = 1.0f / s;
    }
    __syncthreads();

    // ---------------- pass 2: P, O += P*V, buckets ----------------
    v8f oacc = 0.0f;
    for (int j0 = 0; j0 < Ssz; j0 += 64) {
        int jb = j0 + jt * 16;
        int j = jb + lm;
        if (j0 + 64 < Ssz) {
            __builtin_prefetch(kbase + (size_t)(j + 64) * Dk, 0, 1);
            __builtin_prefetch(vtb + (size_t)(jt * 16 + lm) * Ssz + j0 + 64, 0, 1);
        }
        v8f acc = 0.0f;
        #pragma unroll
        for (int kki = 0; kki < 2; ++kki) {
            const bf16_t* krow = kbase + (size_t)j * Dk + kki * 32;
            v16bf bb;
            #pragma unroll
            for (int e = 0; e < 16; ++e) bb[e] = krow[kb2 + e];
            acc = wmma_bf16(aq[kki], bb, acc);
        }
        #pragma unroll
        for (int r = 0; r < 8; ++r) {
            int row = rt * 16 + 8 * hf + r;
            int i = i0 + row;
            int rel = j - i;
            rel = rel < -128 ? -128 : (rel > 128 ? 128 : rel);
            rel += 128;
            float scv = acc[r] + qabase[(size_t)i * RP + rel];
            float p = __expf(scv - mrow[row]);
            Pl[row][jt * 16 + lm] = f2bf(p);
            atomicAdd(&wbuck[row][rel], p);
        }
        __syncthreads();
        // O accumulation: wave -> (row tile rt, v tile jt)
        #pragma unroll
        for (int kki = 0; kki < 2; ++kki) {
            int kk = kki * 32;
            int prow = rt * 16 + lm;
            v16bf a;
            #pragma unroll
            for (int e = 0; e < 8; ++e) {
                a[e]     = Pl[prow][kk + kb2 + e];
                a[8 + e] = Pl[prow][kk + kb2 + 8 + e];
            }
            int vcol = jt * 16 + lm;
            const bf16_t* vrow = vtb + (size_t)vcol * Ssz + j0 + kk;
            v16bf bb;
            #pragma unroll
            for (int e = 0; e < 16; ++e) bb[e] = vrow[kb2 + e];
            oacc = wmma_bf16(a, bb, oacc);
        }
        __syncthreads();
    }

    // normalize O, then z += (wbuck/s) @ a_v via 9 WMMAs
    #pragma unroll
    for (int r = 0; r < 8; ++r) oacc[r] *= rsrow[rt * 16 + 8 * hf + r];

    for (int kt = 0; kt < 9; ++kt) {
        int prow = rt * 16 + lm;
        float rs = rsrow[prow];
        v16bf a;
        #pragma unroll
        for (int e = 0; e < 8; ++e) {
            a[e]     = f2bf(wbuck[prow][kt * 32 + kb2 + e] * rs);
            a[8 + e] = f2bf(wbuck[prow][kt * 32 + kb2 + 8 + e] * rs);
        }
        int vcol = jt * 16 + lm;
        const bf16_t* avrow = avt + (size_t)vcol * RP + kt * 32;
        v16bf bb;
        #pragma unroll
        for (int e = 0; e < 16; ++e) bb[e] = avrow[kb2 + e];
        oacc = wmma_bf16(a, bb, oacc);
    }

    int vcol = jt * 16 + lm;
    #pragma unroll
    for (int r = 0; r < 8; ++r) {
        int row = i0 + rt * 16 + 8 * hf + r;
        Zb[((size_t)bh * Ssz + row) * Dk + vcol] = oacc[r];
    }
}

// =====================================================================
// K4: out[b][s][v] = sum_h w_o[h] * Zb[b*H+h][s][v]
// =====================================================================
__global__ __launch_bounds__(256) void k_reduce(const float* __restrict__ Zb,
                                                const float* __restrict__ w_o,
                                                float* __restrict__ out) {
    int t = blockIdx.x * 256 + threadIdx.x;
    if (t >= Bsz * Ssz * Dk) return;
    int v = t & 63, s = (t >> 6) & (Ssz - 1), b = t >> 16;
    float sum = 0.0f;
    #pragma unroll
    for (int h = 0; h < Hsz; ++h)
        sum += w_o[h] * Zb[((size_t)(b * Hsz + h) * Ssz + s) * Dk + v];
    out[t] = sum;
}

extern "C" void kernel_launch(void* const* d_in, const int* in_sizes, int n_in,
                              void* d_out, int out_size, void* d_ws, size_t ws_size,
                              hipStream_t stream) {
    const float* xs  = (const float*)d_in[0];
    const float* w_q = (const float*)d_in[1];
    const float* w_k = (const float*)d_in[2];
    const float* w_v = (const float*)d_in[3];
    const float* w_o = (const float*)d_in[4];
    const float* a_k = (const float*)d_in[5];
    const float* a_v = (const float*)d_in[6];
    float* out = (float*)d_out;

    char* ws = (char*)d_ws;
    size_t off = 0;
    bf16_t* Qb = (bf16_t*)(ws + off); off += (size_t)BH * Ssz * Dk * 2;   // 4 MB
    bf16_t* Kb = (bf16_t*)(ws + off); off += (size_t)BH * Ssz * Dk * 2;   // 4 MB
    bf16_t* Vt = (bf16_t*)(ws + off); off += (size_t)BH * Ssz * Dk * 2;   // 4 MB
    float* QA = (float*)(ws + off); off += (size_t)BH * Ssz * RP * 4;     // 36 MB
    bf16_t* avt = (bf16_t*)(ws + off); off += ((size_t)Dk * RP * 2 + 255) & ~(size_t)255;
    float* Zb = (float*)(ws + off); off += (size_t)BH * Ssz * Dk * 4;     // 8 MB

    dim3 blk(256);
    k_avt<<<(Dk * RP + 255) / 256, blk, 0, stream>>>(a_v, avt);
    dim3 gproj(BH, Ssz / 64);
    k_proj<false><<<gproj, blk, 0, stream>>>(xs, w_q, Qb, 0.125f);  // Q pre-scaled
    k_proj<false><<<gproj, blk, 0, stream>>>(xs, w_k, Kb, 1.0f);
    k_proj<true ><<<gproj, blk, 0, stream>>>(xs, w_v, Vt, 1.0f);
    k_qa<<<gproj, blk, 0, stream>>>(Qb, a_k, QA);
    dim3 gattn(BH, Ssz / 32);
    k_attn<<<gattn, blk, 0, stream>>>(Qb, Kb, Vt, QA, avt, Zb);
    k_reduce<<<(Bsz * Ssz * Dk + 255) / 256, blk, 0, stream>>>(Zb, w_o, out);
}
